// MSA_cls_10926396801699
// MI455X (gfx1250) — matrix operand order repaired and checked by reading, not compiled, not run on hardware
//
#include <hip/hip_runtime.h>

#define B_    2
#define NTOK  8191
#define NF    8192
#define DIM   512
#define HEADS 8
#define DH    64
#define BH    (B_*HEADS)
#define LM    256
#define LGRP  32
#define KIN   1024
#define QKVW  1536
#define KER   33
#define NLAYERS 2

typedef __attribute__((ext_vector_type(16))) __bf16 v16bf;
typedef __attribute__((ext_vector_type(8)))  float  v8f;
typedef __attribute__((ext_vector_type(4)))  unsigned int v4u;

union FragAB { unsigned int u[8]; v4u u4[2]; v16bf v; };
union FragC  { float f[8]; v8f v; };

#if __has_builtin(__builtin_amdgcn_cvt_pk_bf16_f32)
typedef __attribute__((ext_vector_type(2))) __bf16 v2bf;
__device__ __forceinline__ unsigned int pack2bf(float a, float b) {
  union { v2bf v; unsigned int u; } cv;
  cv.v = __builtin_amdgcn_cvt_pk_bf16_f32(a, b);   // single v_cvt_pk_bf16_f32
  return cv.u;
}
#else
__device__ __forceinline__ unsigned short f2bf(float f) {
  unsigned int u = __float_as_uint(f);
  u += 0x7FFFu + ((u >> 16) & 1u);   // round-to-nearest-even
  return (unsigned short)(u >> 16);
}
__device__ __forceinline__ unsigned int pack2bf(float a, float b) {
  return (unsigned int)f2bf(a) | ((unsigned int)f2bf(b) << 16);
}
#endif

enum { EPI_NONE=0, EPI_BIAS_RELU=1, EPI_SCALE=2, EPI_AIMINUS=3, EPI_ACCUM=4, EPI_ACCUM_BIAS=5 };

#define BM 128
#define BN 128
#define BK 32
#define SAP 20   // padded LDS row stride in dwords (16 kpairs + 4 pad): 16B-aligned, bank-spread
#define SBP 20

// C(M,N) = A(M,K) x B(K,N); A row-major contiguous-k, row stride sa (even);
// B element (k,n) at B[k*sbk + n*sbn]; C row stride sc. blockIdx.z batches.
// K must be a multiple of 32 and M,N clamp-safe (dense allocations) -- true for all call sites.
template<int EPI>
__global__ __launch_bounds__(256)
void gemm_bf16(const float* __restrict__ A, const float* __restrict__ Bm, float* __restrict__ C,
               int M, int N, int K, int sa, int sbk, int sbn, int sc,
               long long ba, long long bb, long long bc,
               const float* __restrict__ bias, float alpha)
{
  __shared__ unsigned int sA32[BM * SAP];  // [m][kpair] packed 2xbf16
  __shared__ unsigned int sB32[BN * SBP];  // [n][kpair] packed 2xbf16
  const int tid = threadIdx.x;
  const int m0 = blockIdx.y * BM;
  const int n0 = blockIdx.x * BN;
  const float* Ab = A + (long long)blockIdx.z * ba;
  const float* Bb = Bm + (long long)blockIdx.z * bb;
  float* Cb = C + (long long)blockIdx.z * bc;

  const int wid = tid >> 5, lane = tid & 31;
  const int wm = (wid & 3) * 32, wn = (wid >> 2) * 64;
  const int l16 = lane & 15, half = lane >> 4;

  v8f acc[2][4] = {};

  for (int k0 = 0; k0 < K; k0 += BK) {
    // ---- stage A tile: 128 rows x 16 kpairs, float2 loads, branch-free clamp ----
    #pragma unroll
    for (int e = 0; e < 8; ++e) {
      int id = e * 256 + tid;
      int m = id >> 4, p = id & 15;
      int gm = m0 + m; if (gm >= M) gm = M - 1;
      const float* ap = Ab + (long long)gm * sa + (k0 + 2 * p);
      float2 v2 = *(const float2*)ap;
      __builtin_prefetch(ap + BK, 0, 3);     // speculative: safe past end (ISA 7.3)
      sA32[m * SAP + p] = pack2bf(v2.x, v2.y);
    }
    // ---- stage B tile: 128 cols x 16 kpairs ----
    #pragma unroll
    for (int e = 0; e < 8; ++e) {
      int id = e * 256 + tid;
      int nn = id & 127, p = id >> 7;
      int gn = n0 + nn; if (gn >= N) gn = N - 1;
      long long kb = (long long)(k0 + 2 * p) * sbk + (long long)gn * sbn;
      __builtin_prefetch(Bb + kb + (long long)BK * sbk, 0, 3);
      sB32[nn * SBP + p] = pack2bf(Bb[kb], Bb[kb + sbk]);
    }
    __syncthreads();

    // ---- fragments via ds_load_b128 (ISA 7.12.2 16-bit A 16x32 / B 32x16 layouts) ----
    FragAB a[2], b[4];
    #pragma unroll
    for (int i = 0; i < 2; ++i) {
      int base = (wm + i * 16 + l16) * SAP + 4 * half;
      a[i].u4[0] = *(const v4u*)&sA32[base];      // kpairs {0..3}+4h  -> VGPR 0..3
      a[i].u4[1] = *(const v4u*)&sA32[base + 8];  // kpairs {8..11}+4h -> VGPR 4..7
    }
    #pragma unroll
    for (int j = 0; j < 4; ++j) {
      int base = (wn + j * 16 + l16) * SBP + 4 * half;
      b[j].u4[0] = *(const v4u*)&sB32[base];      // kpairs {0..3}+4h  -> VGPR 0..3
      b[j].u4[1] = *(const v4u*)&sB32[base + 8];  // kpairs {8..11}+4h -> VGPR 4..7
    }
    #pragma unroll
    for (int i = 0; i < 2; ++i)
      #pragma unroll
      for (int j = 0; j < 4; ++j)
        acc[i][j] = __builtin_amdgcn_wmma_f32_16x16x32_bf16(
            false, a[i].v, false, b[j].v, (short)0, acc[i][j], false, false);
    __syncthreads();
  }

  #pragma unroll
  for (int i = 0; i < 2; ++i) {
    #pragma unroll
    for (int j = 0; j < 4; ++j) {
      FragC r; r.v = acc[i][j];
      int nn = n0 + wn + j * 16 + l16;
      if (nn >= N) continue;
      #pragma unroll
      for (int rr = 0; rr < 8; ++rr) {
        int mm = m0 + wm + i * 16 + rr + 8 * half;
        if (mm >= M) continue;
        long long off = (long long)mm * sc + nn;
        float v = r.f[rr];
        if      (EPI == EPI_NONE)       Cb[off] = v;
        else if (EPI == EPI_BIAS_RELU)  Cb[off] = fmaxf(v + bias[nn], 0.0f);
        else if (EPI == EPI_SCALE)      Cb[off] = v * alpha;
        else if (EPI == EPI_AIMINUS)    Cb[off] = ((mm == nn) ? alpha : 0.0f) - v;
        else if (EPI == EPI_ACCUM)      Cb[off] += v;
        else                            Cb[off] = Cb[off] + v + bias[nn];
      }
    }
  }
}

__global__ __launch_bounds__(256)
void layernorm_rows(const float* __restrict__ in, float* __restrict__ out,
                    const float* __restrict__ g, const float* __restrict__ b)
{
  __shared__ float red[256];
  const float* p = in + (long long)blockIdx.x * DIM;
  float* q = out + (long long)blockIdx.x * DIM;
  int tid = threadIdx.x;
  float v0 = p[tid], v1 = p[tid + 256];
  red[tid] = v0 + v1; __syncthreads();
  for (int st = 128; st > 0; st >>= 1) { if (tid < st) red[tid] += red[tid + st]; __syncthreads(); }
  float mu = red[0] * (1.0f / DIM); __syncthreads();
  float d0 = v0 - mu, d1 = v1 - mu;
  red[tid] = d0 * d0 + d1 * d1; __syncthreads();
  for (int st = 128; st > 0; st >>= 1) { if (tid < st) red[tid] += red[tid + st]; __syncthreads(); }
  float inv = rsqrtf(red[0] * (1.0f / DIM) + 1e-5f);
  q[tid]       = d0 * inv * g[tid]       + b[tid];
  q[tid + 256] = d1 * inv * g[tid + 256] + b[tid + 256];
}

__global__ __launch_bounds__(256)
void softmax_rows(float* __restrict__ d, int cols, long long ld)
{
  __shared__ float red[256];
  float* p = d + (long long)blockIdx.x * ld;
  int tid = threadIdx.x;
  float mx = -3.4e38f;
  for (int c = tid; c < cols; c += 256) mx = fmaxf(mx, p[c]);
  red[tid] = mx; __syncthreads();
  for (int st = 128; st > 0; st >>= 1) { if (tid < st) red[tid] = fmaxf(red[tid], red[tid + st]); __syncthreads(); }
  mx = red[0]; __syncthreads();
  float s = 0.0f;
  for (int c = tid; c < cols; c += 256) s += __expf(p[c] - mx);
  red[tid] = s; __syncthreads();
  for (int st = 128; st > 0; st >>= 1) { if (tid < st) red[tid] += red[tid + st]; __syncthreads(); }
  float inv = 1.0f / red[0];
  for (int c = tid; c < cols; c += 256) p[c] = __expf(p[c] - mx) * inv;
}

__global__ void split_qkv(const float* __restrict__ qkv, float* __restrict__ qh,
                          float* __restrict__ kh, float* __restrict__ vh)
{
  long long idx = (long long)blockIdx.x * 256 + threadIdx.x;
  if (idx >= (long long)B_ * NF * HEADS * DH) return;
  int d = (int)(idx & 63);
  long long t = idx >> 6;
  int hh = (int)(t & 7);
  long long bi = t >> 3;
  int i = (int)(bi % NF), b = (int)(bi / NF);
  long long src = ((long long)(b * NF + i)) * QKVW + hh * DH + d;
  long long dst = (((long long)(b * HEADS + hh)) * NF + i) * DH + d;
  qh[dst] = qkv[src] * 0.125f;           // dh^-0.5
  kh[dst] = qkv[src + 512];
  vh[dst] = qkv[src + 1024];
}

__global__ void landmarks_k(const float* __restrict__ qh, const float* __restrict__ kh,
                            float* __restrict__ ql, float* __restrict__ kl)
{
  int idx = blockIdx.x * 256 + threadIdx.x;          // BH*LM*DH
  if (idx >= BH * LM * DH) return;
  int d = idx & 63, t = idx >> 6;
  int j = t & 255, bh = t >> 8;
  long long base = ((long long)bh * NF + (long long)j * LGRP) * DH + d;
  float sq = 0.0f, sk = 0.0f;
  for (int r = 0; r < LGRP; ++r) { sq += qh[base + (long long)r * DH]; sk += kh[base + (long long)r * DH]; }
  ql[idx] = sq * (1.0f / LGRP);
  kl[idx] = sk * (1.0f / LGRP);
}

__global__ void init_scal(float* scal) { ((int*)scal)[0] = 0; ((int*)scal)[1] = 0; }

__global__ __launch_bounds__(256)
void abssum_max(const float* __restrict__ a2, float* scal, int mode)
{
  __shared__ float red[256];
  int bh = blockIdx.x >> 8, r = blockIdx.x & 255, tid = threadIdx.x;
  const float* base = a2 + (long long)bh * LM * LM;
  float v = (mode == 0) ? base[r * LM + tid] : base[tid * LM + r];
  red[tid] = fabsf(v); __syncthreads();
  for (int st = 128; st > 0; st >>= 1) { if (tid < st) red[tid] += red[tid + st]; __syncthreads(); }
  if (tid == 0) atomicMax((int*)&scal[mode], __float_as_int(red[0]));  // positive floats
}

__global__ void build_z0(const float* __restrict__ a2, const float* __restrict__ scal,
                         float* __restrict__ z)
{
  long long idx = (long long)blockIdx.x * 256 + threadIdx.x;
  if (idx >= (long long)BH * LM * LM) return;
  int j = (int)(idx & 255), i = (int)((idx >> 8) & 255);
  long long bh = idx >> 16;
  float inv = 1.0f / (scal[0] * scal[1]);
  z[idx] = a2[bh * (LM * LM) + (long long)j * LM + i] * inv;   // transpose
}

__global__ void aim_k(const float* __restrict__ src, float* __restrict__ dst, float alpha)
{
  long long idx = (long long)blockIdx.x * 256 + threadIdx.x;
  if (idx >= (long long)BH * LM * LM) return;
  int j = (int)(idx & 255), i = (int)((idx >> 8) & 255);
  dst[idx] = ((i == j) ? alpha : 0.0f) - src[idx];
}

__global__ void dwconv(const float* __restrict__ vh, const float* __restrict__ w,
                       float* __restrict__ outh)
{
  long long idx = (long long)blockIdx.x * 256 + threadIdx.x;   // BH*NF*DH
  if (idx >= (long long)BH * NF * DH) return;
  int d = (int)(idx & 63);
  long long t = idx >> 6;
  int i = (int)(t % NF), bh = (int)(t / NF), hh = bh & 7;
  float s = 0.0f;
  for (int k = 0; k < KER; ++k) {
    int ii = i + k - KER / 2;
    if (ii >= 0 && ii < NF) s += w[hh * KER + k] * vh[((long long)bh * NF + ii) * DH + d];
  }
  outh[idx] = s;
}

__global__ void merge_heads(const float* __restrict__ outh, float* __restrict__ outm)
{
  long long idx = (long long)blockIdx.x * 256 + threadIdx.x;   // = (b*NF+i)*512 + h*64 + d
  if (idx >= (long long)B_ * NF * HEADS * DH) return;
  int d = (int)(idx & 63);
  long long t = idx >> 6;
  int hh = (int)(t & 7);
  long long bi = t >> 3;
  int i = (int)(bi % NF), b = (int)(bi / NF);
  outm[idx] = outh[(((long long)(b * HEADS + hh)) * NF + i) * DH + d];
}

__global__ void write_cls(const float* __restrict__ cls, float* __restrict__ x)
{
  int idx = blockIdx.x * 256 + threadIdx.x;
  if (idx >= B_ * DIM) return;
  int b = idx >> 9, c = idx & 511;
  x[(long long)b * NF * DIM + c] = cls[c];
}

__global__ __launch_bounds__(256)
void final_head(const float* __restrict__ x, const float* __restrict__ g, const float* __restrict__ bb,
                const float* __restrict__ w2, const float* __restrict__ b2, float* __restrict__ out)
{
  __shared__ float red[256];
  int b = blockIdx.x, tid = threadIdx.x;
  const float* p = x + (long long)b * NF * DIM;      // cls row
  float v0 = p[tid], v1 = p[tid + 256];
  red[tid] = v0 + v1; __syncthreads();
  for (int st = 128; st > 0; st >>= 1) { if (tid < st) red[tid] += red[tid + st]; __syncthreads(); }
  float mu = red[0] * (1.0f / DIM); __syncthreads();
  float d0 = v0 - mu, d1 = v1 - mu;
  red[tid] = d0 * d0 + d1 * d1; __syncthreads();
  for (int st = 128; st > 0; st >>= 1) { if (tid < st) red[tid] += red[tid + st]; __syncthreads(); }
  float inv = rsqrtf(red[0] * (1.0f / DIM) + 1e-5f); __syncthreads();
  float l0 = d0 * inv * g[tid] + bb[tid];
  float l1 = d1 * inv * g[tid + 256] + bb[tid + 256];
  for (int c = 0; c < 2; ++c) {
    red[tid] = l0 * w2[tid * 2 + c] + l1 * w2[(tid + 256) * 2 + c]; __syncthreads();
    for (int st = 128; st > 0; st >>= 1) { if (tid < st) red[tid] += red[tid + st]; __syncthreads(); }
    if (tid == 0) out[b * 2 + c] = red[0] + b2[c];
    __syncthreads();
  }
}

static void launch_gemm(int epi, const float* A, const float* B, float* C,
                        int M, int N, int K, int sa, int sbk, int sbn, int sc,
                        long long ba, long long bb, long long bc, int batches,
                        const float* bias, float alpha, hipStream_t s)
{
  dim3 g((N + BN - 1) / BN, (M + BM - 1) / BM, batches), blk(256);
  switch (epi) {
    case EPI_NONE:       gemm_bf16<EPI_NONE>      <<<g, blk, 0, s>>>(A,B,C,M,N,K,sa,sbk,sbn,sc,ba,bb,bc,bias,alpha); break;
    case EPI_BIAS_RELU:  gemm_bf16<EPI_BIAS_RELU> <<<g, blk, 0, s>>>(A,B,C,M,N,K,sa,sbk,sbn,sc,ba,bb,bc,bias,alpha); break;
    case EPI_SCALE:      gemm_bf16<EPI_SCALE>     <<<g, blk, 0, s>>>(A,B,C,M,N,K,sa,sbk,sbn,sc,ba,bb,bc,bias,alpha); break;
    case EPI_AIMINUS:    gemm_bf16<EPI_AIMINUS>   <<<g, blk, 0, s>>>(A,B,C,M,N,K,sa,sbk,sbn,sc,ba,bb,bc,bias,alpha); break;
    case EPI_ACCUM:      gemm_bf16<EPI_ACCUM>     <<<g, blk, 0, s>>>(A,B,C,M,N,K,sa,sbk,sbn,sc,ba,bb,bc,bias,alpha); break;
    default:             gemm_bf16<EPI_ACCUM_BIAS><<<g, blk, 0, s>>>(A,B,C,M,N,K,sa,sbk,sbn,sc,ba,bb,bc,bias,alpha); break;
  }
}

extern "C" void kernel_launch(void* const* d_in, const int* in_sizes, int n_in,
                              void* d_out, int out_size, void* d_ws, size_t ws_size,
                              hipStream_t stream)
{
  (void)in_sizes; (void)n_in; (void)out_size; (void)ws_size;
  const float* h     = (const float*)d_in[0];
  const float* w1    = (const float*)d_in[1];
  const float* b1    = (const float*)d_in[2];
  const float* cls   = (const float*)d_in[3];
  const float* ln_g  = (const float*)d_in[4];
  const float* ln_b  = (const float*)d_in[5];
  const float* w_qkv = (const float*)d_in[6];
  const float* w_out = (const float*)d_in[7];
  const float* b_out = (const float*)d_in[8];
  const float* convw = (const float*)d_in[9];
  const float* fn_g  = (const float*)d_in[10];
  const float* fn_b  = (const float*)d_in[11];
  const float* w2    = (const float*)d_in[12];
  const float* b2    = (const float*)d_in[13];
  float* out = (float*)d_out;

  float* ws = (float*)d_ws;
  size_t off = 0;
  auto alloc = [&](size_t n) { float* p = ws + off; off += n; return p; };
  float* x    = alloc((size_t)B_ * NF * DIM);
  float* lnx  = alloc((size_t)B_ * NF * DIM);
  float* qkv  = alloc((size_t)B_ * NF * QKVW);
  float* qh   = alloc((size_t)BH * NF * DH);     // later reused as outh
  float* kh   = alloc((size_t)BH * NF * DH);     // later reused as outm
  float* vh   = alloc((size_t)BH * NF * DH);
  float* ql   = alloc((size_t)BH * LM * DH);
  float* kl   = alloc((size_t)BH * LM * DH);
  float* sim1 = alloc((size_t)BH * NF * LM);     // attn1
  float* sim3 = alloc((size_t)BH * LM * NF);     // attn3, later t2
  float* a2   = alloc((size_t)BH * LM * LM);     // attn2
  float* za   = alloc((size_t)BH * LM * LM);
  float* zb   = alloc((size_t)BH * LM * LM);
  float* xz   = alloc((size_t)BH * LM * LM);
  float* ta   = alloc((size_t)BH * LM * LM);
  float* tb   = alloc((size_t)BH * LM * LM);
  float* t1   = alloc((size_t)BH * LM * DH);
  float* scal = alloc(16);

  const long long LL = (long long)LM * LM;

  // x[b,1:,:] = relu(h @ w1 + b1)  (batched over b: bs_b = 0)
  launch_gemm(EPI_BIAS_RELU, h, w1, x + DIM, NTOK, DIM, KIN, KIN, DIM, 1, DIM,
              (long long)NTOK * KIN, 0, (long long)NF * DIM, B_, b1, 0.0f, stream);
  write_cls<<<(B_ * DIM + 255) / 256, 256, 0, stream>>>(cls, x);

  for (int L = 0; L < NLAYERS; ++L) {
    layernorm_rows<<<B_ * NF, 256, 0, stream>>>(x, lnx, ln_g + L * DIM, ln_b + L * DIM);
    launch_gemm(EPI_NONE, lnx, w_qkv + (size_t)L * DIM * QKVW, qkv,
                B_ * NF, QKVW, DIM, DIM, QKVW, 1, QKVW, 0, 0, 0, 1, nullptr, 0.0f, stream);
    split_qkv<<<(BH * NF * DH + 255) / 256, 256, 0, stream>>>(qkv, qh, kh, vh);
    landmarks_k<<<(BH * LM * DH + 255) / 256, 256, 0, stream>>>(qh, kh, ql, kl);

    // sim1 = q @ k_l^T : (NF,LM,K=DH), B(k=d,n=j)=kl[j*DH+d]
    launch_gemm(EPI_NONE, qh, kl, sim1, NF, LM, DH, DH, 1, DH, LM,
                (long long)NF * DH, (long long)LM * DH, (long long)NF * LM, BH, nullptr, 0.0f, stream);
    // sim2 = q_l @ k_l^T
    launch_gemm(EPI_NONE, ql, kl, a2, LM, LM, DH, DH, 1, DH, LM,
                (long long)LM * DH, (long long)LM * DH, LL, BH, nullptr, 0.0f, stream);
    // sim3 = q_l @ k^T : B(k=d,n=j)=kh[j*DH+d]
    launch_gemm(EPI_NONE, ql, kh, sim3, LM, NF, DH, DH, 1, DH, NF,
                (long long)LM * DH, (long long)NF * DH, (long long)LM * NF, BH, nullptr, 0.0f, stream);

    softmax_rows<<<BH * NF, 256, 0, stream>>>(sim1, LM, LM);
    softmax_rows<<<BH * LM, 256, 0, stream>>>(a2, LM, LM);
    softmax_rows<<<BH * LM, 256, 0, stream>>>(sim3, NF, NF);

    // pinv(attn2): global col/row scaling, then 6 polynomial iterations
    init_scal<<<1, 1, 0, stream>>>(scal);
    abssum_max<<<BH * LM, 256, 0, stream>>>(a2, scal, 0);
    abssum_max<<<BH * LM, 256, 0, stream>>>(a2, scal, 1);
    build_z0<<<(int)((BH * LL + 255) / 256), 256, 0, stream>>>(a2, scal, za);
    float* zc = za; float* zn = zb;
    for (int it = 0; it < 6; ++it) {
      launch_gemm(EPI_NONE, a2, zc, xz, LM, LM, LM, LM, LM, 1, LM, LL, LL, LL, BH, nullptr, 0.0f, stream);
      aim_k<<<(int)((BH * LL + 255) / 256), 256, 0, stream>>>(xz, ta, 7.0f);
      launch_gemm(EPI_AIMINUS, xz, ta, tb, LM, LM, LM, LM, LM, 1, LM, LL, LL, LL, BH, nullptr, 15.0f, stream);
      launch_gemm(EPI_AIMINUS, xz, tb, ta, LM, LM, LM, LM, LM, 1, LM, LL, LL, LL, BH, nullptr, 13.0f, stream);
      launch_gemm(EPI_SCALE, zc, ta, zn, LM, LM, LM, LM, LM, 1, LM, LL, LL, LL, BH, nullptr, 0.25f, stream);
      float* tmp = zc; zc = zn; zn = tmp;
    }

    // t1 = attn3 @ v : (LM,DH,K=NF), B(k=i,n=d)=vh[i*DH+d]
    launch_gemm(EPI_NONE, sim3, vh, t1, LM, DH, NF, NF, DH, 1, DH,
                (long long)LM * NF, (long long)NF * DH, (long long)LM * DH, BH, nullptr, 0.0f, stream);
    // depthwise conv residual into outh (= qh buffer, q no longer needed)
    dwconv<<<(int)(((long long)BH * NF * DH + 255) / 256), 256, 0, stream>>>(
        vh, convw + (size_t)L * HEADS * KER, qh);
    // t2 = attn1 @ pinv -> reuse sim3 buffer
    launch_gemm(EPI_NONE, sim1, zc, sim3, NF, LM, LM, LM, LM, 1, LM,
                (long long)NF * LM, LL, (long long)NF * LM, BH, nullptr, 0.0f, stream);
    // outh += t2 @ t1
    launch_gemm(EPI_ACCUM, sim3, t1, qh, NF, DH, LM, LM, DH, 1, DH,
                (long long)NF * LM, (long long)LM * DH, (long long)NF * DH, BH, nullptr, 0.0f, stream);
    // (b,h,n,d) -> (b,n,h*d) into outm (= kh buffer)
    merge_heads<<<(int)(((long long)BH * NF * DH + 255) / 256), 256, 0, stream>>>(qh, kh);
    // x += outm @ w_out + b_out
    launch_gemm(EPI_ACCUM_BIAS, kh, w_out + (size_t)L * DIM * DIM, x,
                B_ * NF, DIM, DIM, DIM, DIM, 1, DIM, 0, 0, 0, 1, b_out + L * DIM, 0.0f, stream);
  }

  final_head<<<B_, 256, 0, stream>>>(x, fn_g, fn_b, w2, b2, out);
}
